// ARAPLoss_62929860821306
// MI455X (gfx1250) — matrix operand order repaired and checked by reading, not compiled, not run on hardware
//
#include <hip/hip_runtime.h>
#include <hip/hip_bf16.h>
#include <stdint.h>

// ---------------------------------------------------------------------------
// ARAP loss, MI455X (gfx1250, wave32).
//   sub = L @ [dx_b0|dx_b1|x_b0|x_b1]  via V_WMMA_F32_16X16X32_BF16 (hi/lo
//   bf16 split of B for ~f32 accuracy; L is {0,1} = exact in bf16).
//   Hot loop is branch/atomic-free: nonzeros of L are recorded as a 16-bit
//   mask per lane (exact f32 FMA chain, since L values are {0,1}) into a
//   2 MB bitmap; a separate sparse scan evaluates the loss on edges only.
//   L is streamed exactly once: 64 MB @ 23.3 TB/s => ~2.8 us floor.
// ---------------------------------------------------------------------------

#define NVERT     4096
#define KCHUNKS   128          // 4096 / 32
#define NSLICES   4            // K split across blockIdx.y
#define KC_SLICE  (KCHUNKS / NSLICES)
#define NBMWORDS  (KCHUNKS * NVERT)      // u32 bitmap words: kc*4096 + row

typedef __attribute__((ext_vector_type(16))) __bf16 v16bf;
typedef __attribute__((ext_vector_type(8)))  float  v8f;

union BPack { uint4 u[2]; v16bf v; };   // 32 bytes: one lane's B fragment

// Workspace layout (bytes):
//   0        wsB_hi : [128 kc][32 lane][16 h] bf16     131072
//   131072   wsB_lo : same                             131072
//   262144   subp   : [4 slice][4096 row][16 col] f32  1048576
//   1310720  sub    : [4096 row][16 col] f32           262144
//   1572864  bitmap : [128 kc][4096 row][2 hi] u16     2097152
//   3670016  accum  : [2] f32
//   3670024  count  : u32
#define WS_BLO   131072
#define WS_SUBP  262144
#define WS_SUB   1310720
#define WS_BM    1572864
#define WS_ACC   3670016
#define WS_CNT   3670024

// ---------------------------------------------------------------------------
__global__ void arap_init(unsigned* __restrict__ count, float* __restrict__ accum) {
  if (threadIdx.x == 0) { *count = 0u; accum[0] = 0.0f; accum[1] = 0.0f; }
}

// Pack B = [dx_b0 | dx_b1 | x_b0 | x_b1 | 0...] (4096 x 16) into the exact
// per-lane WMMA B-layout, split into bf16 hi + bf16 residual (lo).
// Index = kc*512 + lane*16 + h ; lane&15 = column n, (lane>>4)*8 + h + (h>=8 ? 8 : 0) = k%32.
__global__ void arap_pack_b(const float* __restrict__ dx, const float* __restrict__ x,
                            __bf16* __restrict__ bhi, __bf16* __restrict__ blo) {
  const int tid  = blockIdx.x * blockDim.x + threadIdx.x;   // 65536 threads
  const int kc   = tid >> 9;
  const int lane = (tid >> 4) & 31;
  const int h    = tid & 15;
  const int n    = lane & 15;
  const int k    = kc * 32 + ((lane >> 4) << 3) + h + (h >= 8 ? 8 : 0);        // vertex index j
  float v = 0.0f;
  if (n < 6)       { const int b = n / 3, d = n % 3;            v = dx[(b * NVERT + k) * 3 + d]; }
  else if (n < 12) { const int m = n - 6, b = m / 3, d = m % 3; v =  x[(b * NVERT + k) * 3 + d]; }
  const __bf16 hibits = (__bf16)v;
  bhi[tid] = hibits;
  blo[tid] = (__bf16)(v - (float)hibits);                   // residual, ~16-bit mantissa total
}

// Main fused kernel: stream L once; WMMA-accumulate sub (per K-slice) and
// emit a nonzero bitmap. No branches, no atomics in the hot loop.
__global__ void __launch_bounds__(64)
arap_gemm_bitmap(const float* __restrict__ L,
                 const __bf16* __restrict__ bhi, const __bf16* __restrict__ blo,
                 float* __restrict__ subp, unsigned short* __restrict__ bm) {
  const int lane  = threadIdx.x & 31;
  const int wave  = threadIdx.x >> 5;
  const int lo    = lane & 15;           // A row within tile / C column
  const int hi    = lane >> 4;           // K-half selector per ISA A layout
  const int tile  = blockIdx.x * 2 + wave;        // 0..255  (16-row M tile)
  const int slice = blockIdx.y;                   // 0..3    (K slice)
  const int row   = tile * 16 + lo;               // global row of L this lane owns
  const float* __restrict__ arow = L + (size_t)row * NVERT;
  const uint4* __restrict__ bh4 = (const uint4*)bhi;   // 64 uint4 per kc
  const uint4* __restrict__ bl4 = (const uint4*)blo;

  v8f accH = {0.f,0.f,0.f,0.f,0.f,0.f,0.f,0.f};
  v8f accL = {0.f,0.f,0.f,0.f,0.f,0.f,0.f,0.f};

  const int kc0 = slice * KC_SLICE;
  for (int kc = kc0; kc < kc0 + KC_SLICE; ++kc) {
    const int kbase = kc * 32;
    // A fragment: 16x32 f32 tile in ISA A layout -> two contiguous 8-float
    // runs per lane; hi=0/hi=1 lanes of a row jointly cover each 128B line.
    const float4 f0 = *(const float4*)(arow + kbase + hi * 8);
    const float4 f1 = *(const float4*)(arow + kbase + hi * 8 + 4);
    const float4 f2 = *(const float4*)(arow + kbase + hi * 8 + 16);
    const float4 f3 = *(const float4*)(arow + kbase + hi * 8 + 20);
    if (kc + 1 < kc0 + KC_SLICE)
      __builtin_prefetch(arow + kbase + 32 + hi * 8, 0, 1);  // global_prefetch_b8

    const float v[16] = {f0.x,f0.y,f0.z,f0.w, f1.x,f1.y,f1.z,f1.w,
                         f2.x,f2.y,f2.z,f2.w, f3.x,f3.y,f3.z,f3.w};
    v16bf a;
    #pragma unroll
    for (int t = 0; t < 16; ++t) a[t] = (__bf16)v[t];        // {0,1}: exact

    // Nonzero mask: L values are exactly 0.0/1.0, so sum v[t]*2^t is an
    // exact integer in f32 (<= 65535 < 2^24). Pure FMA chain, no compares.
    float mf = 0.0f;
    #pragma unroll
    for (int t = 0; t < 16; ++t)
      mf = __builtin_fmaf(v[t], (float)(1u << t), mf);
    const unsigned short mask = (unsigned short)(unsigned)mf;
    // Wave writes one contiguous 64B line: bm[kc][tile*16+lo][hi].
    bm[(size_t)kc * (NVERT * 2) + (size_t)row * 2 + hi] = mask;

    BPack bH, bL;
    bH.u[0] = bh4[kc * 64 + lane * 2 + 0];
    bH.u[1] = bh4[kc * 64 + lane * 2 + 1];
    bL.u[0] = bl4[kc * 64 + lane * 2 + 0];
    bL.u[1] = bl4[kc * 64 + lane * 2 + 1];

    accH = __builtin_amdgcn_wmma_f32_16x16x32_bf16(false, a, false, bH.v,
                                                   (short)0, accH, false, false);
    accL = __builtin_amdgcn_wmma_f32_16x16x32_bf16(false, a, false, bL.v,
                                                   (short)0, accL, false, false);
  }

  // C layout: VGPR r -> row tile*16 + hi*8 + r, column lo. Sum hi+lo parts.
  float* __restrict__ srow =
      subp + (((size_t)slice * NVERT) + (size_t)tile * 16 + hi * 8) * 16 + lo;
  #pragma unroll
  for (int r = 0; r < 8; ++r) srow[(size_t)r * 16] = accH[r] + accL[r];
}

// Collapse the 4 K-slice partial sums: sub = sum_s subp[s].
__global__ void arap_reduce_sub(const float* __restrict__ subp, float* __restrict__ sub) {
  const int t = blockIdx.x * blockDim.x + threadIdx.x;      // 65536 threads
  float s = 0.0f;
  #pragma unroll
  for (int sl = 0; sl < NSLICES; ++sl) s += subp[(size_t)sl * (NVERT * 16) + t];
  sub[t] = s;
}

// Sparse scan of the bitmap: per set bit compute
// | ||x_sub[j]-x[i]||^2 - ||dx_sub[j]-dx[i]||^2 | per batch; popcount -> n_edges.
__global__ void arap_edge_scan(const float* __restrict__ dx, const float* __restrict__ x,
                               const float* __restrict__ sub,
                               const unsigned* __restrict__ bm32,
                               float* __restrict__ accum, unsigned* __restrict__ count) {
  float s0 = 0.0f, s1 = 0.0f;
  unsigned cnt = 0;
  for (unsigned w = blockIdx.x * blockDim.x + threadIdx.x; w < (unsigned)NBMWORDS;
       w += gridDim.x * blockDim.x) {
    unsigned m = bm32[w];                 // two u16 masks: [15:0]=hi0, [31:16]=hi1
    if (!m) continue;                     // ~91% of words
    cnt += (unsigned)__popc(m);
    const int kc    = (int)(w >> 12);     // w = kc*4096 + row
    const int i     = (int)(w & 4095u);   // L row
    const int kbase = kc * 32;
    // orig[i] for both batches
    const float dx0x = dx[i * 3 + 0], dx0y = dx[i * 3 + 1], dx0z = dx[i * 3 + 2];
    const float dx1x = dx[(NVERT + i) * 3 + 0], dx1y = dx[(NVERT + i) * 3 + 1],
                dx1z = dx[(NVERT + i) * 3 + 2];
    const float x0x = x[i * 3 + 0], x0y = x[i * 3 + 1], x0z = x[i * 3 + 2];
    const float x1x = x[(NVERT + i) * 3 + 0], x1y = x[(NVERT + i) * 3 + 1],
                x1z = x[(NVERT + i) * 3 + 2];
    while (m) {
      const int p = __builtin_ctz(m);
      m &= m - 1;
      const int half = p >> 4;            // which u16 (hi)
      const int t    = p & 15;            // bit within lane mask
      const int j    = kbase + half * 8 + t + (t >= 8 ? 8 : 0);
      const float4 sa = *(const float4*)(sub + (size_t)j * 16);      // cols 0-3
      const float4 sb = *(const float4*)(sub + (size_t)j * 16 + 4);  // cols 4-7
      const float4 sc = *(const float4*)(sub + (size_t)j * 16 + 8);  // cols 8-11
      // cols: 0-2 dx_b0, 3-5 dx_b1, 6-8 x_b0, 9-11 x_b1  (sub indexed by j)
      float a0 = sa.x - dx0x, a1 = sa.y - dx0y, a2 = sa.z - dx0z;
      const float ddx0 = a0 * a0 + a1 * a1 + a2 * a2;
      a0 = sa.w - dx1x; a1 = sb.x - dx1y; a2 = sb.y - dx1z;
      const float ddx1 = a0 * a0 + a1 * a1 + a2 * a2;
      a0 = sb.z - x0x; a1 = sb.w - x0y; a2 = sc.x - x0z;
      const float dxx0 = a0 * a0 + a1 * a1 + a2 * a2;
      a0 = sc.y - x1x; a1 = sc.z - x1y; a2 = sc.w - x1z;
      const float dxx1 = a0 * a0 + a1 * a1 + a2 * a2;
      s0 += fabsf(dxx0 - ddx0);
      s1 += fabsf(dxx1 - ddx1);
    }
  }
  #pragma unroll
  for (int off = 16; off > 0; off >>= 1) {
    s0  += __shfl_down(s0, off, 32);
    s1  += __shfl_down(s1, off, 32);
    cnt += __shfl_down(cnt, off, 32);
  }
  if ((threadIdx.x & 31) == 0) {
    atomicAdd(&accum[0], s0);
    atomicAdd(&accum[1], s1);
    atomicAdd(count, cnt);
  }
}

__global__ void arap_finalize(const float* __restrict__ accum,
                              const unsigned* __restrict__ count,
                              float* __restrict__ out) {
  if (threadIdx.x == 0) {
    const unsigned c = *count;
    const float n = (float)(c > 0u ? c : 1u);
    out[0] = accum[0] / n;
    out[1] = accum[1] / n;
  }
}

// ---------------------------------------------------------------------------
extern "C" void kernel_launch(void* const* d_in, const int* in_sizes, int n_in,
                              void* d_out, int out_size, void* d_ws, size_t ws_size,
                              hipStream_t stream) {
  const float* dx = (const float*)d_in[0];   // [2, 4096, 3]
  const float* x  = (const float*)d_in[1];   // [2, 4096, 3]
  const float* L  = (const float*)d_in[2];   // [4096, 4096] {0,1}
  float* out = (float*)d_out;                // [2]

  char* ws = (char*)d_ws;
  __bf16*         wsBhi = (__bf16*)(ws);
  __bf16*         wsBlo = (__bf16*)(ws + WS_BLO);
  float*          subp  = (float*)(ws + WS_SUBP);
  float*          sub   = (float*)(ws + WS_SUB);
  unsigned short* bm    = (unsigned short*)(ws + WS_BM);
  float*          accum = (float*)(ws + WS_ACC);
  unsigned*       count = (unsigned*)(ws + WS_CNT);

  arap_init<<<1, 32, 0, stream>>>(count, accum);
  arap_pack_b<<<256, 256, 0, stream>>>(dx, x, wsBhi, wsBlo);        // 65536 threads
  dim3 grid(128, NSLICES);                                          // 1024 waves
  arap_gemm_bitmap<<<grid, 64, 0, stream>>>(L, wsBhi, wsBlo, subp, bm);
  arap_reduce_sub<<<256, 256, 0, stream>>>(subp, sub);              // 65536 threads
  arap_edge_scan<<<128, 256, 0, stream>>>(dx, x, sub, (const unsigned*)bm,
                                          accum, count);
  arap_finalize<<<1, 32, 0, stream>>>(accum, count, out);
}
